// TimeMixKV_55662776156514
// MI455X (gfx1250) — matrix operand order repaired and checked by reading, not compiled, not run on hardware
//
#include <hip/hip_runtime.h>
#include <hip/hip_bf16.h>
#include <math.h>

typedef __attribute__((ext_vector_type(16))) _Float16 v16h;
typedef __attribute__((ext_vector_type(8)))  _Float16 v8h;
typedef __attribute__((ext_vector_type(8)))  float    v8f;
typedef __attribute__((ext_vector_type(4)))  int      v4i;

static constexpr int CB  = 2048;   // channel dim C
static constexpr int BB  = 4;      // batch
static constexpr int TT  = 2048;   // time
static constexpr int BMt = 128;    // block tile M
static constexpr int BNt = 128;    // block tile N
static constexpr int BKt = 32;     // k-step == one WMMA K

// ---- CDNA5 async global->LDS copy (ASYNCcnt path), with safe fallback ------
#if defined(__has_builtin)
#if __has_builtin(__builtin_amdgcn_global_load_async_to_lds_b128)
#define HAVE_ASYNC_COPY 1
#endif
#if __has_builtin(__builtin_amdgcn_s_wait_asynccnt)
#define HAVE_WAIT_ASYNC 1
#endif
#endif

#if defined(HAVE_ASYNC_COPY)
typedef __attribute__((address_space(1))) v4i as1_v4i;
typedef __attribute__((address_space(3))) v4i as3_v4i;
__device__ __forceinline__ void cp16_g2l(void* lds, const void* g) {
    __builtin_amdgcn_global_load_async_to_lds_b128((as1_v4i*)g, (as3_v4i*)lds, 0, 0);
}
__device__ __forceinline__ void wait_copies() {
#if defined(HAVE_WAIT_ASYNC)
    __builtin_amdgcn_s_wait_asynccnt(0);
#else
    asm volatile("s_wait_asynccnt 0" ::: "memory");
#endif
}
#else
__device__ __forceinline__ void cp16_g2l(void* lds, const void* g) {
    *(uint4*)lds = *(const uint4*)g;          // load + ds_store fallback
}
__device__ __forceinline__ void wait_copies() {} // __syncthreads covers DScnt
#endif

// ---------------------------------------------------------------------------
// f32 -> f16 elementwise convert (for weights)
// ---------------------------------------------------------------------------
__global__ __launch_bounds__(256)
void cvt_f32_f16_kernel(const float* __restrict__ src, _Float16* __restrict__ dst, int n) {
    int i = blockIdx.x * 256 + threadIdx.x;
    if (i < n) dst[i] = (_Float16)src[i];
}

// ---------------------------------------------------------------------------
// LayerNorm over C, writes f16 activations. One block per (b,t) row.
// ---------------------------------------------------------------------------
__global__ __launch_bounds__(256)
void ln_kernel(const float* __restrict__ x, const float* __restrict__ ln_w,
               const float* __restrict__ ln_b, _Float16* __restrict__ xn_h) {
    const int row = blockIdx.x;
    const float* xr = x + (size_t)row * CB;
    const int tid = threadIdx.x;

    float s = 0.f, s2 = 0.f;
    for (int i = tid; i < CB; i += 256) { float v = xr[i]; s += v; s2 += v * v; }

    __shared__ float rs[256], rq[256];
    rs[tid] = s; rq[tid] = s2;
    __syncthreads();
    for (int st = 128; st > 0; st >>= 1) {
        if (tid < st) { rs[tid] += rs[tid + st]; rq[tid] += rq[tid + st]; }
        __syncthreads();
    }
    const float mu  = rs[0] * (1.0f / CB);
    const float var = rq[0] * (1.0f / CB) - mu * mu;
    const float inv = rsqrtf(var + 1e-5f);

    _Float16* orow = xn_h + (size_t)row * CB;
    for (int i = tid; i < CB; i += 256)
        orow[i] = (_Float16)((xr[i] - mu) * inv * ln_w[i] + ln_b[i]);
}

// ---------------------------------------------------------------------------
// WMMA GEMM core: out[m,d] = sum_c A[m,c]*W[d,c], block tile 128x128, BK=32,
// 8 waves as 4(M) x 2(N); each wave owns 32x64 = 8 accumulators (8 WMMA/k-step).
// LDS double-buffered; staging via async global->LDS copies when available.
// mode: 0 = exp(clip(v+bias,-10,10)), 1 = identity, 2 = sigmoid(v+bias)
// ---------------------------------------------------------------------------
__device__ __forceinline__
void gemm_core(const _Float16* __restrict__ A, const _Float16* __restrict__ W,
               float* __restrict__ out, const float* __restrict__ bias,
               int m0, int n0, int mode,
               _Float16 (&As)[2][BMt][BKt], _Float16 (&Bs)[2][BNt][BKt]) {
    const int tid  = threadIdx.x;
    const int lane = tid & 31;
    const int l15  = lane & 15;
    const int hsel = lane >> 4;       // 0/1 half-wave select
    const int wv   = tid >> 5;        // 0..7
    const int mw   = wv >> 1;         // 0..3 : wave row (32 rows each)
    const int nw   = wv & 1;          // 0..1 : wave col (64 cols each)

    // staging map: idx in [0,512) -> row=idx>>2, col=(idx&3)*8 (16B vectors)
    const int sr0 = tid >> 2,          sc0 = (tid & 3) * 8;
    const int sr1 = (tid + 256) >> 2,  sc1 = sc0;

    v8f acc[2][4] = {};

    auto stage = [&](int buf, int k0) {
        cp16_g2l(&As[buf][sr0][sc0], &A[(size_t)(m0 + sr0) * CB + k0 + sc0]);
        cp16_g2l(&As[buf][sr1][sc1], &A[(size_t)(m0 + sr1) * CB + k0 + sc1]);
        cp16_g2l(&Bs[buf][sr0][sc0], &W[(size_t)(n0 + sr0) * CB + k0 + sc0]);
        cp16_g2l(&Bs[buf][sr1][sc1], &W[(size_t)(n0 + sr1) * CB + k0 + sc1]);
        if (k0 + 2 * BKt < CB)        // speculative prefetch two tiles ahead
            __builtin_prefetch(&W[(size_t)(n0 + sr0) * CB + k0 + 2 * BKt + sc0], 0, 0);
    };

    stage(0, 0);
    wait_copies();
    __syncthreads();

    int p = 0;
    for (int k0 = 0; k0 < CB; k0 += BKt) {
        const bool has_next = (k0 + BKt) < CB;
        if (has_next) stage(p ^ 1, k0 + BKt);

        // A fragments (16x32 wave32 layout): elems 0..15 -> k=hsel*16+e contiguous
        union AF { v16h v; v8h h[2]; } a[2];
        #pragma unroll
        for (int i = 0; i < 2; ++i) {
            const int r = mw * 32 + i * 16 + l15;
            a[i].v = *(const v16h*)&As[p][r][hsel * 16];
        }
        // B fragments (32x16): lane col N=l15, k = hsel*16 + 0..15 contiguous
        v16h b[4];
        #pragma unroll
        for (int j = 0; j < 4; ++j)
            b[j] = *(const v16h*)&Bs[p][nw * 64 + j * 16 + l15][hsel * 16];

        #pragma unroll
        for (int i = 0; i < 2; ++i)
            #pragma unroll
            for (int j = 0; j < 4; ++j)
                acc[i][j] = __builtin_amdgcn_wmma_f32_16x16x32_f16(
                    false, a[i].v, false, b[j], (short)0, acc[i][j], false, false);

        if (has_next) {
            wait_copies();
            __syncthreads();
            p ^= 1;
        }
    }

    // epilogue: C/D layout — lane col N=l15, VGPR v -> row M = v + 8*hsel
    #pragma unroll
    for (int i = 0; i < 2; ++i) {
        const int mbase = m0 + mw * 32 + i * 16 + 8 * hsel;
        #pragma unroll
        for (int j = 0; j < 4; ++j) {
            const int d  = n0 + nw * 64 + j * 16 + l15;
            const float bd = bias ? bias[d] : 0.0f;
            #pragma unroll
            for (int v = 0; v < 8; ++v) {
                float val = acc[i][j][v];
                if (mode == 0)      val = __expf(fminf(fmaxf(val + bd, -10.f), 10.f));
                else if (mode == 2) val = 1.0f / (1.0f + __expf(-(val + bd)));
                out[(size_t)(mbase + v) * CB + d] = val;
            }
        }
    }
}

// fused K/V/R projections: grid.y in [0, 3*C/128); weight selected by column band
__global__ __launch_bounds__(256)
void gemm_kvr_kernel(const _Float16* __restrict__ A,
                     const _Float16* __restrict__ Wk, const _Float16* __restrict__ Wv,
                     const _Float16* __restrict__ Wr,
                     float* __restrict__ outk, float* __restrict__ outv,
                     float* __restrict__ outr,
                     const float* __restrict__ bonus, const float* __restrict__ br) {
    __shared__ alignas(64) _Float16 As[2][BMt][BKt];   // 16 KB
    __shared__ alignas(64) _Float16 Bs[2][BNt][BKt];   // 16 KB
    const int wsel = blockIdx.y >> 4;                  // C/128 == 16 bands/weight
    const int n0   = (blockIdx.y & 15) * BNt;
    const int m0   = blockIdx.x * BMt;
    const _Float16* W   = (wsel == 0) ? Wk   : (wsel == 1) ? Wv   : Wr;
    float*          out = (wsel == 0) ? outk : (wsel == 1) ? outv : outr;
    const float*    bs  = (wsel == 0) ? bonus : (wsel == 2) ? br : nullptr;
    const int       md  = wsel;                        // 0:exp-clip 1:id 2:sigmoid
    gemm_core(A, W, out, bs, m0, n0, md, As, Bs);
}

// single projection (output GEMM)
__global__ __launch_bounds__(256)
void gemm_one_kernel(const _Float16* __restrict__ A, const _Float16* __restrict__ W,
                     float* __restrict__ out) {
    __shared__ alignas(64) _Float16 As[2][BMt][BKt];
    __shared__ alignas(64) _Float16 Bs[2][BNt][BKt];
    gemm_core(A, W, out, nullptr, blockIdx.x * BMt, blockIdx.y * BNt, 1, As, Bs);
}

// ---------------------------------------------------------------------------
// Sequential decay scan over T; one thread per (b,c) channel, coalesced in c.
// decay = exp(-softplus(w)) = 1/(1+exp(w))
// ---------------------------------------------------------------------------
__global__ __launch_bounds__(256)
void scan_kernel(const float* __restrict__ ek, const float* __restrict__ vv,
                 const float* __restrict__ rr, const float* __restrict__ time_decay,
                 _Float16* __restrict__ y_h) {
    const int idx = blockIdx.x * 256 + threadIdx.x;   // 0 .. B*C-1
    const int c = idx & (CB - 1);
    const int b = idx >> 11;
    const float decay = 1.0f / (1.0f + __expf(time_decay[c]));

    float a = 0.f, den = 0.f;
    size_t o = (size_t)b * TT * CB + c;
    for (int t = 0; t < TT; ++t, o += CB) {
        const float e = ek[o];
        const float v = vv[o];
        a   = fmaf(decay, a,   e * v);
        den = fmaf(decay, den, e);
        y_h[o] = (_Float16)(rr[o] * (a / (den + 1e-6f)));
    }
}

// ---------------------------------------------------------------------------
// Launch
// ---------------------------------------------------------------------------
extern "C" void kernel_launch(void* const* d_in, const int* in_sizes, int n_in,
                              void* d_out, int out_size, void* d_ws, size_t ws_size,
                              hipStream_t stream) {
    (void)in_sizes; (void)n_in; (void)out_size; (void)ws_size;

    const float* x          = (const float*)d_in[0];
    const float* ln_w       = (const float*)d_in[1];
    const float* ln_b       = (const float*)d_in[2];
    const float* Wk         = (const float*)d_in[3];
    const float* Wv         = (const float*)d_in[4];
    const float* Wr         = (const float*)d_in[5];
    const float* br         = (const float*)d_in[6];
    const float* time_decay = (const float*)d_in[7];
    const float* time_bonus = (const float*)d_in[8];
    const float* Wout       = (const float*)d_in[9];
    float* out = (float*)d_out;

    const size_t BTC = (size_t)BB * TT * CB;
    const size_t CC  = (size_t)CB * CB;

    _Float16* xn_h = (_Float16*)d_ws;
    _Float16* wk_h = xn_h + BTC;
    _Float16* wv_h = wk_h + CC;
    _Float16* wr_h = wv_h + CC;
    _Float16* wo_h = wr_h + CC;
    _Float16* y_h  = wo_h + CC;
    float*    ekb  = (float*)(y_h + BTC);
    float*    vvb  = ekb + BTC;
    float*    rrb  = vvb + BTC;

    const int M = BB * TT;   // 8192

    const int cvb = (int)((CC + 255) / 256);
    cvt_f32_f16_kernel<<<cvb, 256, 0, stream>>>(Wk,   wk_h, (int)CC);
    cvt_f32_f16_kernel<<<cvb, 256, 0, stream>>>(Wv,   wv_h, (int)CC);
    cvt_f32_f16_kernel<<<cvb, 256, 0, stream>>>(Wr,   wr_h, (int)CC);
    cvt_f32_f16_kernel<<<cvb, 256, 0, stream>>>(Wout, wo_h, (int)CC);

    ln_kernel<<<M, 256, 0, stream>>>(x, ln_w, ln_b, xn_h);

    dim3 gf(M / BMt, 3 * (CB / BNt));   // (64, 48)
    gemm_kvr_kernel<<<gf, 256, 0, stream>>>(xn_h, wk_h, wv_h, wr_h,
                                            ekb, vvb, rrb, time_bonus, br);

    scan_kernel<<<(BB * CB) / 256, 256, 0, stream>>>(ekb, vvb, rrb, time_decay, y_h);

    dim3 go(M / BMt, CB / BNt);         // (64, 16)
    gemm_one_kernel<<<go, 256, 0, stream>>>(y_h, wo_h, out);
}
